// DualPathMSA_17892833755590
// MI455X (gfx1250) — matrix operand order repaired and checked by reading, not compiled, not run on hardware
//
#include <hip/hip_runtime.h>
#include <hip/hip_bf16.h>
#include <math.h>

// DualPathMSA for MI455X (gfx1250, wave32, WMMA).
// All GEMMs use v_wmma_f32_16x16x32_bf16 with f32 accumulation and 2x2
// register blocking (each wave owns a 32x32 output macro-tile -> 4 WMMAs per
// 4 fragment loads per K-step).
// Required workspace: ~870 MB (carved from d_ws below).

typedef __bf16 bf16_t;
typedef __attribute__((ext_vector_type(16))) __bf16 v16bf;
typedef __attribute__((ext_vector_type(8)))  __bf16 v8bf;
typedef __attribute__((ext_vector_type(8)))  float  v8f;

#define DEVINL __device__ __forceinline__

static constexpr int kT  = 1024;   // sequence length
static constexpr int kD  = 768;    // model dim
static constexpr int kH  = 12;     // heads
static constexpr int kDK = 64;     // head dim
static constexpr int kBH = 48;     // batch * heads
static constexpr int kM  = 4096;   // batch * seq

// ---------------------------------------------------------------------------
// WMMA fragment loaders (wave32; A and B use the same contiguous k order).
//
// A (16x32 bf16, M x K): lane l holds row m = l&15. hi = l>>4.
//   elements 0..15 = K hi*16 .. hi*16+15
DEVINL v16bf wmma_load_a(const bf16_t* __restrict__ base, int ld, int m0, int k0) {
  const int l  = threadIdx.x & 31;
  const int hi = (l >> 4) & 1;
  const bf16_t* row = base + (size_t)(m0 + (l & 15)) * ld + k0 + hi * 16;
  v8bf c0 = *(const v8bf*)(row);
  v8bf c1 = *(const v8bf*)(row + 8);
  v16bf a;
#pragma unroll
  for (int i = 0; i < 8; ++i) { a[i] = c0[i]; a[i + 8] = c1[i]; }
  return a;
}

// B (32x16 bf16, K x N), supplied as B^T row-major ("W" with shape [N, K]):
// lane l holds column n = l&15, K = (l>>4)*16 + i -> one 32-byte read.
DEVINL v16bf wmma_load_b(const bf16_t* __restrict__ wt, int ld, int n0, int k0) {
  const int l = threadIdx.x & 31;
  const bf16_t* p = wt + (size_t)(n0 + (l & 15)) * ld + k0 + ((l >> 4) << 4);
  return *(const v16bf*)p;
}

DEVINL v8f wmma_acc(v16bf a, v16bf b, v8f c) {
  return __builtin_amdgcn_wmma_f32_16x16x32_bf16(false, a, false, b, (short)0, c,
                                                 false, false);
}

// C/D 16x16 f32: lane l -> column n = l&15; VGPR r -> row (l>>4)*8 + r.
DEVINL void cd_coords(int& mbase, int& ncol) {
  const int l = threadIdx.x & 31;
  ncol  = l & 15;
  mbase = (l >> 4) * 8;
}

// 2x2 register-blocked GEMM inner loop: wave computes C[m0:m0+32, n0:n0+32].
DEVINL void wmma_tile2x2(const bf16_t* __restrict__ A, int lda, int m0,
                         const bf16_t* __restrict__ Wt, int ldb, int n0,
                         int K, v8f (&acc)[2][2]) {
  for (int kk = 0; kk < K; kk += 32) {
    v16bf a0 = wmma_load_a(A, lda, m0,      kk);
    v16bf a1 = wmma_load_a(A, lda, m0 + 16, kk);
    v16bf b0 = wmma_load_b(Wt, ldb, n0,      kk);
    v16bf b1 = wmma_load_b(Wt, ldb, n0 + 16, kk);
    acc[0][0] = wmma_acc(a0, b0, acc[0][0]);
    acc[0][1] = wmma_acc(a0, b1, acc[0][1]);
    acc[1][0] = wmma_acc(a1, b0, acc[1][0]);
    acc[1][1] = wmma_acc(a1, b1, acc[1][1]);
  }
}

// ---------------------------------------------------------------------------
// Block reductions (blockDim.x == 256)
DEVINL float block_reduce_max(float v, float* red) {
  const int t = threadIdx.x;
  red[t] = v; __syncthreads();
  for (int s = 128; s > 0; s >>= 1) {
    if (t < s) red[t] = fmaxf(red[t], red[t + s]);
    __syncthreads();
  }
  float r = red[0]; __syncthreads();
  return r;
}
DEVINL float block_reduce_sum(float v, float* red) {
  const int t = threadIdx.x;
  red[t] = v; __syncthreads();
  for (int s = 128; s > 0; s >>= 1) {
    if (t < s) red[t] += red[t + s];
    __syncthreads();
  }
  float r = red[0]; __syncthreads();
  return r;
}

// ---------------------------------------------------------------------------
// 1. f32 -> bf16 conversion
__global__ __launch_bounds__(256) void k_cvt_bf16(const float* __restrict__ in,
                                                  bf16_t* __restrict__ out, int n) {
  int i = blockIdx.x * 256 + threadIdx.x;
  if (i < n) out[i] = (bf16_t)in[i];
}

// 2. QKV projection: Y = X @ W^T, scattered into head-major q, k and v^T.
//    X: [4096, 768] bf16, W: [2304, 768] bf16.
//    grid (9, 128), block 256: 8 waves x 32x32 macro-tiles; 72 n-macros.
__global__ __launch_bounds__(256) void k_qkv(const bf16_t* __restrict__ X,
                                             const bf16_t* __restrict__ W,
                                             bf16_t* __restrict__ q,
                                             bf16_t* __restrict__ k,
                                             bf16_t* __restrict__ vT) {
  const int wave = threadIdx.x >> 5;
  const int m0 = blockIdx.y * 32;
  const int n0 = (blockIdx.x * 8 + wave) * 32;
  v8f acc[2][2] = {};
  wmma_tile2x2(X, kD, m0, W, kD, n0, kD, acc);

  int mb, nc; cd_coords(mb, nc);
#pragma unroll
  for (int ni = 0; ni < 2; ++ni) {
    const int n = n0 + ni * 16 + nc;
    const int slot = n / kD, within = n % kD;
    const int h = within / kDK, d = within % kDK;
#pragma unroll
    for (int mi = 0; mi < 2; ++mi) {
#pragma unroll
      for (int r = 0; r < 8; ++r) {
        const int m = m0 + mi * 16 + mb + r;
        const int b = m >> 10, t = m & (kT - 1);
        const int bh = b * kH + h;
        const bf16_t val = (bf16_t)acc[mi][ni][r];
        if (slot == 0)      q [((size_t)bh * kT + t) * kDK + d] = val;
        else if (slot == 1) k [((size_t)bh * kT + t) * kDK + d] = val;
        else                vT[((size_t)bh * kDK + d) * kT + t] = val;
      }
    }
  }
}

// 3. Scores: S = scale * q @ k^T per (b,h). grid (4, 32, 48), block 256.
__global__ __launch_bounds__(256) void k_scores(const bf16_t* __restrict__ q,
                                                const bf16_t* __restrict__ kmat,
                                                float* __restrict__ S, float scale) {
  const int bh = blockIdx.z;
  const bf16_t* Q = q    + (size_t)bh * kT * kDK;
  const bf16_t* K = kmat + (size_t)bh * kT * kDK;
  float* Sp = S + (size_t)bh * kT * kT;
  const int wave = threadIdx.x >> 5;
  const int m0 = blockIdx.y * 32;
  const int n0 = (blockIdx.x * 8 + wave) * 32;
  v8f acc[2][2] = {};
  wmma_tile2x2(Q, kDK, m0, K, kDK, n0, kDK, acc);
  int mb, nc; cd_coords(mb, nc);
#pragma unroll
  for (int mi = 0; mi < 2; ++mi)
#pragma unroll
    for (int ni = 0; ni < 2; ++ni)
#pragma unroll
      for (int r = 0; r < 8; ++r)
        Sp[(size_t)(m0 + mi * 16 + mb + r) * kT + n0 + ni * 16 + nc] =
            acc[mi][ni][r] * scale;
}

// 4. Dual softmax + Smix. Row per block. grid (1024, 48), block 256.
//    Smix = S1 + S2 + 0.5*(logaddexp(S1,S2) - S1) - 0.125*S2, written over S1.
__global__ __launch_bounds__(256) void k_softmax_dual(float* __restrict__ S1,
                                                      const float* __restrict__ S2,
                                                      bf16_t* __restrict__ A1,
                                                      bf16_t* __restrict__ A2) {
  __shared__ float red[256];
  const size_t off = ((size_t)blockIdx.y * kT + blockIdx.x) * kT;
  float* s1 = S1 + off; const float* s2 = S2 + off;
  bf16_t* a1 = A1 + off; bf16_t* a2 = A2 + off;

  float v1[4], v2[4];
  float mx1 = -3.0e38f, mx2 = -3.0e38f;
#pragma unroll
  for (int i = 0; i < 4; ++i) {
    const int idx = threadIdx.x + i * 256;
    v1[i] = s1[idx]; v2[i] = s2[idx];
    mx1 = fmaxf(mx1, v1[i]); mx2 = fmaxf(mx2, v2[i]);
  }
  mx1 = block_reduce_max(mx1, red);
  mx2 = block_reduce_max(mx2, red);
  float sm1 = 0.f, sm2 = 0.f;
#pragma unroll
  for (int i = 0; i < 4; ++i) { sm1 += expf(v1[i] - mx1); sm2 += expf(v2[i] - mx2); }
  sm1 = block_reduce_sum(sm1, red);
  sm2 = block_reduce_sum(sm2, red);
  const float inv1 = 1.f / sm1, inv2 = 1.f / sm2;
#pragma unroll
  for (int i = 0; i < 4; ++i) {
    const int idx = threadIdx.x + i * 256;
    a1[idx] = (bf16_t)(expf(v1[i] - mx1) * inv1);
    a2[idx] = (bf16_t)(expf(v2[i] - mx2) * inv2);
    const float lae = fmaxf(v1[i], v2[i]) + log1pf(expf(-fabsf(v1[i] - v2[i])));
    s1[idx] = v1[i] + v2[i] + 0.5f * (lae - v1[i]) - 0.125f * v2[i];
  }
}

// 5. LDS-tiled bf16 transpose (A2 -> A2^T). grid (32, 32, 48), block 256.
__global__ __launch_bounds__(256) void k_transpose(const bf16_t* __restrict__ A,
                                                   bf16_t* __restrict__ AT) {
  __shared__ bf16_t tile[32][33];
  const size_t off = (size_t)blockIdx.z * kT * kT;
  const bf16_t* Ain = A + off;
  bf16_t* Aout = AT + off;
  const int r0 = blockIdx.y * 32, c0 = blockIdx.x * 32;
  const int tx = threadIdx.x & 31, ty = threadIdx.x >> 5;  // 32 x 8
#pragma unroll
  for (int i = 0; i < 32; i += 8)
    tile[ty + i][tx] = Ain[(size_t)(r0 + ty + i) * kT + c0 + tx];
  __syncthreads();
#pragma unroll
  for (int i = 0; i < 32; i += 8)
    Aout[(size_t)(c0 + ty + i) * kT + r0 + tx] = tile[tx][ty + i];
}

// 6. Chain: C = A1 @ A2 (K=1024), Smix += 0.5*log(C + eps). grid (4, 32, 48).
__global__ __launch_bounds__(256) void k_chain(const bf16_t* __restrict__ A1,
                                               const bf16_t* __restrict__ A2T,
                                               float* __restrict__ Smix) {
  const int bh = blockIdx.z;
  const bf16_t* Ap = A1  + (size_t)bh * kT * kT;
  const bf16_t* Bt = A2T + (size_t)bh * kT * kT;
  float* Sp = Smix + (size_t)bh * kT * kT;
  const int wave = threadIdx.x >> 5;
  const int m0 = blockIdx.y * 32;
  const int n0 = (blockIdx.x * 8 + wave) * 32;
  const int l = threadIdx.x & 31;
  v8f acc[2][2] = {};
  for (int kk = 0; kk < kT; kk += 32) {
    if (kk + 32 < kT) {  // prefetch next B chunks -> global_prefetch_b8
      __builtin_prefetch(Bt + (size_t)(n0 + (l & 15)) * kT + kk + 32, 0, 1);
      __builtin_prefetch(Bt + (size_t)(n0 + 16 + (l & 15)) * kT + kk + 32, 0, 1);
    }
    v16bf a0 = wmma_load_a(Ap, kT, m0,      kk);
    v16bf a1 = wmma_load_a(Ap, kT, m0 + 16, kk);
    v16bf b0 = wmma_load_b(Bt, kT, n0,      kk);
    v16bf b1 = wmma_load_b(Bt, kT, n0 + 16, kk);
    acc[0][0] = wmma_acc(a0, b0, acc[0][0]);
    acc[0][1] = wmma_acc(a0, b1, acc[0][1]);
    acc[1][0] = wmma_acc(a1, b0, acc[1][0]);
    acc[1][1] = wmma_acc(a1, b1, acc[1][1]);
  }
  int mb, nc; cd_coords(mb, nc);
#pragma unroll
  for (int mi = 0; mi < 2; ++mi)
#pragma unroll
    for (int ni = 0; ni < 2; ++ni)
#pragma unroll
      for (int r = 0; r < 8; ++r) {
        const size_t idx =
            (size_t)(m0 + mi * 16 + mb + r) * kT + n0 + ni * 16 + nc;
        Sp[idx] += 0.5f * logf(acc[mi][ni][r] + 1e-6f);
      }
}

// 7. Final softmax: A = softmax(Smix) -> bf16. grid (1024, 48), block 256.
__global__ __launch_bounds__(256) void k_softmax_single(const float* __restrict__ S,
                                                        bf16_t* __restrict__ A) {
  __shared__ float red[256];
  const size_t off = ((size_t)blockIdx.y * kT + blockIdx.x) * kT;
  const float* s = S + off;
  bf16_t* a = A + off;
  float v[4], mx = -3.0e38f;
#pragma unroll
  for (int i = 0; i < 4; ++i) {
    v[i] = s[threadIdx.x + i * 256];
    mx = fmaxf(mx, v[i]);
  }
  mx = block_reduce_max(mx, red);
  float sm = 0.f;
#pragma unroll
  for (int i = 0; i < 4; ++i) sm += expf(v[i] - mx);
  sm = block_reduce_sum(sm, red);
  const float inv = 1.f / sm;
#pragma unroll
  for (int i = 0; i < 4; ++i)
    a[threadIdx.x + i * 256] = (bf16_t)(expf(v[i] - mx) * inv);
}

// 8. t2^T = (A2 @ v2)^T per (b,h). grid (32, 48), block 64 (2 waves = d macro-tiles).
__global__ __launch_bounds__(64) void k_t2(const bf16_t* __restrict__ A2,
                                           const bf16_t* __restrict__ v2T,
                                           bf16_t* __restrict__ t2T) {
  const int bh = blockIdx.y;
  const bf16_t* Ap = A2  + (size_t)bh * kT * kT;
  const bf16_t* Vt = v2T + (size_t)bh * kDK * kT;
  bf16_t* Tt = t2T + (size_t)bh * kDK * kT;
  const int wave = threadIdx.x >> 5;
  const int m0 = blockIdx.x * 32;
  const int n0 = wave * 32;
  v8f acc[2][2] = {};
  wmma_tile2x2(Ap, kT, m0, Vt, kT, n0, kT, acc);
  int mb, nc; cd_coords(mb, nc);
#pragma unroll
  for (int mi = 0; mi < 2; ++mi)
#pragma unroll
    for (int ni = 0; ni < 2; ++ni)
#pragma unroll
      for (int r = 0; r < 8; ++r)  // transposed store
        Tt[(size_t)(n0 + ni * 16 + nc) * kT + m0 + mi * 16 + mb + r] =
            (bf16_t)acc[mi][ni][r];
}

// 9. y = A @ v1 + sigmoid(logit) * (A1 @ t2), merged to [B*T, 768] bf16.
//    grid (32, 48), block 64.
__global__ __launch_bounds__(64) void k_y(const bf16_t* __restrict__ A,
                                          const bf16_t* __restrict__ A1,
                                          const bf16_t* __restrict__ v1T,
                                          const bf16_t* __restrict__ t2T,
                                          const float* __restrict__ logit,
                                          bf16_t* __restrict__ ym) {
  const int bh = blockIdx.y;
  const int b = bh / kH, h = bh % kH;
  const bf16_t* Ap  = A   + (size_t)bh * kT * kT;
  const bf16_t* A1p = A1  + (size_t)bh * kT * kT;
  const bf16_t* Vt  = v1T + (size_t)bh * kDK * kT;
  const bf16_t* Tt  = t2T + (size_t)bh * kDK * kT;
  const int wave = threadIdx.x >> 5;
  const int m0 = blockIdx.x * 32;
  const int n0 = wave * 32;
  v8f acc1[2][2] = {}, acc2[2][2] = {};
  for (int kk = 0; kk < kT; kk += 32) {
    v16bf a0 = wmma_load_a(Ap,  kT, m0,      kk);
    v16bf a1 = wmma_load_a(Ap,  kT, m0 + 16, kk);
    v16bf c0 = wmma_load_a(A1p, kT, m0,      kk);
    v16bf c1 = wmma_load_a(A1p, kT, m0 + 16, kk);
    v16bf bv0 = wmma_load_b(Vt, kT, n0,      kk);
    v16bf bv1 = wmma_load_b(Vt, kT, n0 + 16, kk);
    v16bf bt0 = wmma_load_b(Tt, kT, n0,      kk);
    v16bf bt1 = wmma_load_b(Tt, kT, n0 + 16, kk);
    acc1[0][0] = wmma_acc(a0, bv0, acc1[0][0]);
    acc1[0][1] = wmma_acc(a0, bv1, acc1[0][1]);
    acc1[1][0] = wmma_acc(a1, bv0, acc1[1][0]);
    acc1[1][1] = wmma_acc(a1, bv1, acc1[1][1]);
    acc2[0][0] = wmma_acc(c0, bt0, acc2[0][0]);
    acc2[0][1] = wmma_acc(c0, bt1, acc2[0][1]);
    acc2[1][0] = wmma_acc(c1, bt0, acc2[1][0]);
    acc2[1][1] = wmma_acc(c1, bt1, acc2[1][1]);
  }
  const float w = 1.f / (1.f + expf(-logit[0]));
  int mb, nc; cd_coords(mb, nc);
#pragma unroll
  for (int mi = 0; mi < 2; ++mi)
#pragma unroll
    for (int ni = 0; ni < 2; ++ni)
#pragma unroll
      for (int r = 0; r < 8; ++r) {
        const int m = m0 + mi * 16 + mb + r;
        ym[(size_t)(b * kT + m) * kD + h * kDK + n0 + ni * 16 + nc] =
            (bf16_t)(acc1[mi][ni][r] + w * acc2[mi][ni][r]);
      }
}

// 10. Output projection: out = ym @ proj_w^T -> f32. grid (3, 128), block 256.
__global__ __launch_bounds__(256) void k_proj(const bf16_t* __restrict__ ym,
                                              const bf16_t* __restrict__ wp,
                                              float* __restrict__ out) {
  const int wave = threadIdx.x >> 5;
  const int m0 = blockIdx.y * 32;
  const int n0 = (blockIdx.x * 8 + wave) * 32;
  v8f acc[2][2] = {};
  wmma_tile2x2(ym, kD, m0, wp, kD, n0, kD, acc);
  int mb, nc; cd_coords(mb, nc);
#pragma unroll
  for (int mi = 0; mi < 2; ++mi)
#pragma unroll
    for (int ni = 0; ni < 2; ++ni)
#pragma unroll
      for (int r = 0; r < 8; ++r)
        out[(size_t)(m0 + mi * 16 + mb + r) * kD + n0 + ni * 16 + nc] =
            acc[mi][ni][r];
}

// ---------------------------------------------------------------------------
extern "C" void kernel_launch(void* const* d_in, const int* in_sizes, int n_in,
                              void* d_out, int out_size, void* d_ws, size_t ws_size,
                              hipStream_t stream) {
  const float* x      = (const float*)d_in[0];   // [4,1024,768]
  const float* w1     = (const float*)d_in[1];   // [2304,768]
  const float* w2     = (const float*)d_in[2];   // [2304,768]
  const float* wp     = (const float*)d_in[3];   // [768,768]
  const float* logit  = (const float*)d_in[4];   // [1]
  float* out = (float*)d_out;

  uint8_t* ws = (uint8_t*)d_ws;
  size_t off = 0;
  auto carve = [&](size_t bytes) -> void* {
    off = (off + 255) & ~(size_t)255;
    void* p = ws + off;
    off += bytes;
    return p;
  };

  const size_t nX  = (size_t)kM * kD;          // 3,145,728
  const size_t nW  = (size_t)3 * kD * kD;      // 1,769,472
  const size_t nQ  = (size_t)kBH * kT * kDK;   // 3,145,728
  const size_t nTT = (size_t)kBH * kT * kT;    // 50,331,648

  bf16_t* xb  = (bf16_t*)carve(nX * 2);
  bf16_t* w1b = (bf16_t*)carve(nW * 2);
  bf16_t* w2b = (bf16_t*)carve(nW * 2);
  bf16_t* wpb = (bf16_t*)carve((size_t)kD * kD * 2);
  bf16_t* q1  = (bf16_t*)carve(nQ * 2);
  bf16_t* k1  = (bf16_t*)carve(nQ * 2);
  bf16_t* v1T = (bf16_t*)carve(nQ * 2);
  bf16_t* q2  = (bf16_t*)carve(nQ * 2);
  bf16_t* k2  = (bf16_t*)carve(nQ * 2);
  bf16_t* v2T = (bf16_t*)carve(nQ * 2);
  float*  S1  = (float*)carve(nTT * 4);        // becomes Smix in place
  float*  S2  = (float*)carve(nTT * 4);
  bf16_t* A1  = (bf16_t*)carve(nTT * 2);
  bf16_t* A2  = (bf16_t*)carve(nTT * 2);
  bf16_t* A2T = (bf16_t*)carve(nTT * 2);
  bf16_t* Af  = (bf16_t*)carve(nTT * 2);
  bf16_t* t2T = (bf16_t*)carve(nQ * 2);
  bf16_t* ymg = (bf16_t*)carve(nX * 2);
  (void)ws_size; (void)in_sizes; (void)n_in; (void)out_size;

  // 1. bf16 conversions
  k_cvt_bf16<<<(int)((nX + 255) / 256), 256, 0, stream>>>(x, xb, (int)nX);
  k_cvt_bf16<<<(int)((nW + 255) / 256), 256, 0, stream>>>(w1, w1b, (int)nW);
  k_cvt_bf16<<<(int)((nW + 255) / 256), 256, 0, stream>>>(w2, w2b, (int)nW);
  k_cvt_bf16<<<(int)((kD * kD + 255) / 256), 256, 0, stream>>>(wp, wpb, kD * kD);

  // 2. QKV projections (72 n-macros x 128 m-macros of 32x32)
  k_qkv<<<dim3(9, 128), 256, 0, stream>>>(xb, w1b, q1, k1, v1T);
  k_qkv<<<dim3(9, 128), 256, 0, stream>>>(xb, w2b, q2, k2, v2T);

  // 3. Scores (scale = 1/sqrt(64))
  k_scores<<<dim3(4, 32, 48), 256, 0, stream>>>(q1, k1, S1, 0.125f);
  k_scores<<<dim3(4, 32, 48), 256, 0, stream>>>(q2, k2, S2, 0.125f);

  // 4. Dual softmax + Smix (Smix overwrites S1)
  k_softmax_dual<<<dim3(1024, 48), 256, 0, stream>>>(S1, S2, A1, A2);

  // 5. Transpose A2 for chain GEMM B-operand
  k_transpose<<<dim3(32, 32, 48), 256, 0, stream>>>(A2, A2T);

  // 6. Chain composition + log-fuse into Smix (the 103-GFLOP loop)
  k_chain<<<dim3(4, 32, 48), 256, 0, stream>>>(A1, A2T, S1);

  // 7. Final softmax
  k_softmax_single<<<dim3(1024, 48), 256, 0, stream>>>(S1, Af);

  // 8. t2^T = (A2 @ v2)^T
  k_t2<<<dim3(32, 48), 64, 0, stream>>>(A2, v2T, t2T);

  // 9. y = A @ v1 + sigmoid(logit) * (A1 @ t2) -> merged heads
  k_y<<<dim3(32, 48), 64, 0, stream>>>(Af, A1, v1T, t2T, logit, ymg);

  // 10. Output projection
  k_proj<<<dim3(3, 128), 256, 0, stream>>>(ymg, wpb, out);
}
